// LTCNetwork_36670430773607
// MI455X (gfx1250) — matrix operand order repaired and checked by reading, not compiled, not run on hardware
//
#include <hip/hip_runtime.h>
#include <hip/hip_bf16.h>

// Problem dims (match reference)
#define SEQ    256
#define BATCH  256
#define INDIM  512
#define HID    2048
#define OUTDIM 512

typedef __bf16 bf16;
typedef __bf16 v16bf __attribute__((ext_vector_type(16)));
typedef __bf16 v8bf  __attribute__((ext_vector_type(8)));
typedef float  v8f   __attribute__((ext_vector_type(8)));
typedef float  v4f   __attribute__((ext_vector_type(4)));

static __device__ inline v16bf cat8(v8bf lo, v8bf hi) {
  v16bf r;
#pragma unroll
  for (int i = 0; i < 8; ++i) { r[i] = lo[i]; r[i + 8] = hi[i]; }
  return r;
}

static __device__ inline v8f wmma_bf16(v16bf a, v16bf b, v8f c) {
  return __builtin_amdgcn_wmma_f32_16x16x32_bf16(false, a, false, b, (short)0, c,
                                                 false, false);
}

// CDNA5 async global->LDS copy of 64 contiguous bytes (4 x b128).
// Per ISA 15.18, INST_OFFSET is added to both the LDS and the global address,
// i.e. the same immediate offset advances both sides.
static __device__ inline void async_copy64(unsigned lds_addr, const void* gaddr) {
  unsigned long long ga = (unsigned long long)(uintptr_t)gaddr;
  asm volatile("global_load_async_to_lds_b128 %0, %1, off"
               :: "v"(lds_addr), "v"(ga) : "memory");
  asm volatile("global_load_async_to_lds_b128 %0, %1, off offset:16"
               :: "v"(lds_addr), "v"(ga) : "memory");
  asm volatile("global_load_async_to_lds_b128 %0, %1, off offset:32"
               :: "v"(lds_addr), "v"(ga) : "memory");
  asm volatile("global_load_async_to_lds_b128 %0, %1, off offset:48"
               :: "v"(lds_addr), "v"(ga) : "memory");
}

static __device__ inline void wait_async0() {
  asm volatile("s_wait_asynccnt 0x0" ::: "memory");
}

// ---------------------------------------------------------------------------
// Global-direct WMMA strip (used by the small input/output GEMMs).
//   A: row-major [M x K] bf16; B: row-major [N x K] bf16 (weights) -> C = A*B^T
// Fragment loads map 1:1 onto the CDNA5 16-bit A(16x32)/B(32x16) VGPR layouts.
// ---------------------------------------------------------------------------
template <int NT>
static __device__ inline void gemm_strip_bf16(const bf16* __restrict__ A, int lda,
                                              const bf16* __restrict__ B, int ldb,
                                              int mbase, int nbase, int K,
                                              v8f acc[NT]) {
  const int lane = threadIdx.x & 31;
  const int lh   = lane >> 4;
  const int l16  = lane & 15;
  const bf16* arow = A + (size_t)(mbase + l16) * lda;
  for (int k = 0; k < K; k += 32) {
    v8bf a0 = *(const v8bf*)(arow + k + lh * 8);
    v8bf a1 = *(const v8bf*)(arow + k + 16 + lh * 8);
    v16bf af = cat8(a0, a1);
#pragma unroll
    for (int tt = 0; tt < NT; ++tt) {
      const bf16* brow = B + (size_t)(nbase + tt * 16 + l16) * ldb + k + lh * 8;
      v16bf bfr = cat8(*(const v8bf*)(brow), *(const v8bf*)(brow + 16));
      acc[tt] = wmma_bf16(af, bfr, acc[tt]);
    }
  }
}

// Same, with fp32 A converted to bf16 on the fly (used for x_t).
template <int NT>
static __device__ inline void gemm_strip_f32a(const float* __restrict__ A, int lda,
                                              const bf16* __restrict__ B, int ldb,
                                              int mbase, int nbase, int K,
                                              v8f acc[NT]) {
  const int lane = threadIdx.x & 31;
  const int lh   = lane >> 4;
  const int l16  = lane & 15;
  const float* arow = A + (size_t)(mbase + l16) * lda;
  for (int k = 0; k < K; k += 32) {
    v4f f0 = *(const v4f*)(arow + k + lh * 8);
    v4f f1 = *(const v4f*)(arow + k + lh * 8 + 4);
    v4f f2 = *(const v4f*)(arow + k + 16 + lh * 8);
    v4f f3 = *(const v4f*)(arow + k + 16 + lh * 8 + 4);
    v16bf af;
#pragma unroll
    for (int i = 0; i < 4; ++i) {
      af[i]      = (bf16)f0[i];
      af[i + 4]  = (bf16)f1[i];
      af[i + 8]  = (bf16)f2[i];
      af[i + 12] = (bf16)f3[i];
    }
#pragma unroll
    for (int tt = 0; tt < NT; ++tt) {
      const bf16* brow = B + (size_t)(nbase + tt * 16 + l16) * ldb + k + lh * 8;
      v16bf bfr = cat8(*(const v8bf*)(brow), *(const v8bf*)(brow + 16));
      acc[tt] = wmma_bf16(af, bfr, acc[tt]);
    }
  }
}

// ----------------------------- prep kernels --------------------------------
__global__ __launch_bounds__(256) void k_cvt_bf16(const float* __restrict__ s,
                                                  bf16* __restrict__ d, int n) {
  int i = blockIdx.x * blockDim.x + threadIdx.x;
  if (i < n) d[i] = (bf16)s[i];
}

__global__ __launch_bounds__(256) void k_prep_vec(const float* __restrict__ tau_log,
                                                  const float* __restrict__ b_in,
                                                  const float* __restrict__ b_rec,
                                                  const float* __restrict__ bias,
                                                  float* __restrict__ invtau,
                                                  float* __restrict__ ubias) {
  int i = blockIdx.x * blockDim.x + threadIdx.x;
  if (i < HID) {
    float tv = expf(tau_log[i]);
    tv = fminf(fmaxf(tv, 0.1f), 10.0f);
    invtau[i] = 1.0f / tv;
    ubias[i]  = b_in[i] + b_rec[i] + bias[i];
  }
}

__global__ __launch_bounds__(256) void k_zero_state(float* __restrict__ hf,
                                                    float* __restrict__ htf,
                                                    bf16* __restrict__ htb, int n) {
  int i = blockIdx.x * blockDim.x + threadIdx.x;
  if (i < n) { hf[i] = 0.0f; htf[i] = 0.0f; htb[i] = (bf16)0.0f; }
}

__global__ __launch_bounds__(256) void k_copy(const float* __restrict__ s,
                                              float* __restrict__ d, int n) {
  int i = blockIdx.x * blockDim.x + threadIdx.x;
  if (i < n) d[i] = s[i];
}

// ----------------------- u = x_t @ w_in^T + ubias --------------------------
__global__ __launch_bounds__(256) void k_in_gemm(const float* __restrict__ x,
                                                 const bf16* __restrict__ win,
                                                 const float* __restrict__ ubias,
                                                 float* __restrict__ u) {
  const int w = threadIdx.x >> 5;
  const int wm = w >> 1, wn = w & 1;
  const int mbase = blockIdx.y * 64 + wm * 16;
  const int nbase = blockIdx.x * 128 + wn * 64;
  v8f acc[4];
#pragma unroll
  for (int i = 0; i < 4; ++i)
#pragma unroll
    for (int j = 0; j < 8; ++j) acc[i][j] = 0.0f;
  gemm_strip_f32a<4>(x, INDIM, win, INDIM, mbase, nbase, INDIM, acc);
  const int lane = threadIdx.x & 31, lh = lane >> 4, l16 = lane & 15;
#pragma unroll
  for (int tt = 0; tt < 4; ++tt)
#pragma unroll
    for (int r = 0; r < 8; ++r) {
      int m = mbase + lh * 8 + r;
      int n = nbase + tt * 16 + l16;
      u[(size_t)m * HID + n] = acc[tt][r] + ubias[n];
    }
}

// ---------------------------------------------------------------------------
// Fused RHS eval: z = htmp @ w_rec^T ; RK4 epilogue.
// LDS-tiled, double-buffered via async global->LDS loads.
// Block = 256 thr = 8 waves (4 M-groups x 2 N-groups), tile 128x128, BK=32.
// Wave computes 32x64 (2 A-frags x 4 B-frags, register B reuse x2).
// ---------------------------------------------------------------------------
#define NKC (HID / 32)
__global__ __launch_bounds__(256) void k_rhs(const bf16* __restrict__ htb_in,
                                             const float* __restrict__ htf_in,
                                             float* __restrict__ hf,
                                             const float* __restrict__ u,
                                             const bf16* __restrict__ wrec,
                                             const float* __restrict__ invtau,
                                             float* __restrict__ ksum,
                                             float* __restrict__ htf_out,
                                             bf16* __restrict__ htb_out,
                                             const float* __restrict__ tseq,
                                             int t, int stage) {
  // Buffers: [2][ A:128 rows x 64B | B:128 rows x 64B ] = 2 x 16 KB
  __shared__ __attribute__((aligned(16))) unsigned char smem[2][16384];

  const int tid  = threadIdx.x;
  const int w    = tid >> 5;
  const int wm   = w >> 1;          // 0..3  (M group)
  const int wn   = w & 1;           // 0..1  (N group)
  const int lane = tid & 31, lh = lane >> 4, l16 = lane & 15;
  const int mblk = blockIdx.y * 128;
  const int nblk = blockIdx.x * 128;

  // Cooperative fill: each thread owns one 64-byte row-chunk per K-chunk.
  // tid < 128 -> A row (htmp), tid >= 128 -> B row (w_rec). The LDS image is
  // A rows at [0,8192), B rows at [8192,16384) inside each buffer.
  const unsigned char* grow =
      (tid < 128)
          ? (const unsigned char*)(htb_in + (size_t)(mblk + tid) * HID)
          : (const unsigned char*)(wrec + (size_t)(nblk + (tid - 128)) * HID);
  const unsigned lds_row =
      (unsigned)(uintptr_t)(&smem[0][0]) + (unsigned)tid * 64u;

  // Prologue: chunk 0 -> buffer 0
  async_copy64(lds_row, grow);
  wait_async0();
  __syncthreads();

  v8f acc[2][4];
#pragma unroll
  for (int f = 0; f < 2; ++f)
#pragma unroll
    for (int i = 0; i < 4; ++i)
#pragma unroll
      for (int j = 0; j < 8; ++j) acc[f][i][j] = 0.0f;

  for (int kc = 0; kc < NKC; ++kc) {
    const int cur = kc & 1;
    // Fill the other buffer with chunk kc+1; it completes during this chunk's
    // compute phase (wait+barrier at the bottom of the loop).
    if (kc + 1 < NKC)
      async_copy64(lds_row + (unsigned)(cur ^ 1) * 16384u,
                   grow + (size_t)(kc + 1) * 64);

    const unsigned char* sA = &smem[cur][0];
    const unsigned char* sB = &smem[cur][8192];

    v16bf af[2];
#pragma unroll
    for (int f = 0; f < 2; ++f) {
      const unsigned char* p = sA + ((wm * 32 + f * 16 + l16) * 64 + lh * 16);
      af[f] = cat8(*(const v8bf*)p, *(const v8bf*)(p + 32));
    }
#pragma unroll
    for (int tt = 0; tt < 4; ++tt) {
      const unsigned char* p = sB + ((wn * 64 + tt * 16 + l16) * 64 + lh * 16);
      v16bf bfr = cat8(*(const v8bf*)p, *(const v8bf*)(p + 32));
      acc[0][tt] = wmma_bf16(af[0], bfr, acc[0][tt]);
      acc[1][tt] = wmma_bf16(af[1], bfr, acc[1][tt]);
    }
    wait_async0();
    __syncthreads();
  }

  float dt  = (t < SEQ - 1) ? (tseq[t + 1] - tseq[t]) : 0.1f;
  float hdt = dt * 0.25f;  // dt / N_SUB
#pragma unroll
  for (int f = 0; f < 2; ++f)
#pragma unroll
    for (int tt = 0; tt < 4; ++tt)
#pragma unroll
      for (int r = 0; r < 8; ++r) {
        int m = mblk + wm * 32 + f * 16 + lh * 8 + r;
        int n = nblk + wn * 64 + tt * 16 + l16;
        size_t idx = (size_t)m * HID + n;
        float z  = u[idx] + acc[f][tt][r];
        float kk = (tanhf(z) - htf_in[idx]) * invtau[n];  // rhs()
        float hv = hf[idx];
        if (stage == 3) {
          float hn = hv + (hdt * (1.0f / 6.0f)) * (ksum[idx] + kk);
          hf[idx]      = hn;
          htf_out[idx] = hn;
          htb_out[idx] = (bf16)hn;
        } else {
          float ks = (stage == 0) ? kk : (ksum[idx] + 2.0f * kk);
          ksum[idx] = ks;
          float c  = (stage == 2) ? hdt : 0.5f * hdt;
          float hn = hv + c * kk;
          htf_out[idx] = hn;
          htb_out[idx] = (bf16)hn;
        }
      }
}

// --------------------- out = h @ w_out^T + b_out ---------------------------
__global__ __launch_bounds__(256) void k_out_gemm(const bf16* __restrict__ h,
                                                  const bf16* __restrict__ wout,
                                                  const float* __restrict__ bout,
                                                  float* __restrict__ out) {
  const int w = threadIdx.x >> 5;
  const int wm = w >> 1, wn = w & 1;
  const int mbase = blockIdx.y * 64 + wm * 16;
  const int nbase = blockIdx.x * 128 + wn * 64;
  v8f acc[4];
#pragma unroll
  for (int i = 0; i < 4; ++i)
#pragma unroll
    for (int j = 0; j < 8; ++j) acc[i][j] = 0.0f;
  gemm_strip_bf16<4>(h, HID, wout, HID, mbase, nbase, HID, acc);
  const int lane = threadIdx.x & 31, lh = lane >> 4, l16 = lane & 15;
#pragma unroll
  for (int tt = 0; tt < 4; ++tt)
#pragma unroll
    for (int r = 0; r < 8; ++r) {
      int m = mbase + lh * 8 + r;
      int n = nbase + tt * 16 + l16;
      out[(size_t)m * OUTDIM + n] = acc[tt][r] + bout[n];
    }
}

// ---------------------------------------------------------------------------
extern "C" void kernel_launch(void* const* d_in, const int* in_sizes, int n_in,
                              void* d_out, int out_size, void* d_ws, size_t ws_size,
                              hipStream_t stream) {
  (void)in_sizes; (void)n_in; (void)out_size; (void)ws_size;
  const float* x     = (const float*)d_in[0];
  const float* tseq  = (const float*)d_in[1];
  const float* w_in  = (const float*)d_in[2];
  const float* b_in  = (const float*)d_in[3];
  const float* w_rec = (const float*)d_in[4];
  const float* b_rec = (const float*)d_in[5];
  const float* tau   = (const float*)d_in[6];
  const float* bias  = (const float*)d_in[7];
  const float* w_out = (const float*)d_in[8];
  const float* b_out = (const float*)d_in[9];
  float* out = (float*)d_out;

  char* ws = (char*)d_ws;
  size_t off = 0;
  auto take = [&](size_t bytes) -> char* {
    char* p = ws + off;
    off = (off + bytes + 255) & ~(size_t)255;
    return p;
  };
  bf16*  wrec_bf = (bf16*)take((size_t)HID * HID * 2);
  bf16*  win_bf  = (bf16*)take((size_t)HID * INDIM * 2);
  bf16*  wout_bf = (bf16*)take((size_t)OUTDIM * HID * 2);
  float* invtau  = (float*)take((size_t)HID * 4);
  float* ubias   = (float*)take((size_t)HID * 4);
  float* u       = (float*)take((size_t)BATCH * HID * 4);
  float* hf      = (float*)take((size_t)BATCH * HID * 4);
  float* htf0    = (float*)take((size_t)BATCH * HID * 4);
  float* htf1    = (float*)take((size_t)BATCH * HID * 4);
  bf16*  htb0    = (bf16*)take((size_t)BATCH * HID * 2);
  bf16*  htb1    = (bf16*)take((size_t)BATCH * HID * 2);
  float* ksum    = (float*)take((size_t)BATCH * HID * 4);
  float* htf[2]  = {htf0, htf1};
  bf16*  htb[2]  = {htb0, htb1};

  int nrec = HID * HID;
  k_cvt_bf16<<<(nrec + 255) / 256, 256, 0, stream>>>(w_rec, wrec_bf, nrec);
  int nin = HID * INDIM;
  k_cvt_bf16<<<(nin + 255) / 256, 256, 0, stream>>>(w_in, win_bf, nin);
  int nout = OUTDIM * HID;
  k_cvt_bf16<<<(nout + 255) / 256, 256, 0, stream>>>(w_out, wout_bf, nout);
  k_prep_vec<<<(HID + 255) / 256, 256, 0, stream>>>(tau, b_in, b_rec, bias, invtau, ubias);
  int ns = BATCH * HID;
  k_zero_state<<<(ns + 255) / 256, 256, 0, stream>>>(hf, htf[0], htb[0], ns);

  dim3 blk(256);
  dim3 gin(HID / 128, BATCH / 64);      // 16 x 4
  dim3 grh(HID / 128, BATCH / 128);     // 16 x 2
  dim3 gout(OUTDIM / 128, BATCH / 64);  // 4 x 4

  int cur = 0;
  for (int t = 0; t < SEQ; ++t) {
    k_in_gemm<<<gin, blk, 0, stream>>>(x + (size_t)t * BATCH * INDIM, win_bf, ubias, u);
    for (int sub = 0; sub < 4; ++sub) {
      for (int st = 0; st < 4; ++st) {
        k_rhs<<<grh, blk, 0, stream>>>(htb[cur], htf[cur], hf, u, wrec_bf, invtau,
                                       ksum, htf[cur ^ 1], htb[cur ^ 1], tseq, t, st);
        cur ^= 1;
      }
    }
    k_out_gemm<<<gout, blk, 0, stream>>>(htb[cur], wout_bf, b_out,
                                         out + (size_t)t * BATCH * OUTDIM);
  }
  k_copy<<<(ns + 255) / 256, 256, 0, stream>>>(hf, out + (size_t)SEQ * BATCH * OUTDIM, ns);
}
